// GNNBranch_82772609729193
// MI455X (gfx1250) — compile-verified, matrix operand order checked by reading, not run on hardware
//
#include <hip/hip_runtime.h>
#include <hip/hip_bf16.h>

// ---------------------------------------------------------------- constants
#define NNODES   50000
#define NEDGES   500000
#define NGRAPHS  32
#define ESM      1280
#define LIGD     36
#define LIGK     64     // ligand K padded to multiple of 32 (zero-padded B rows)
#define HID      256
#define HEADS    4
#define ATTD     128
#define ATTH     4
#define NEGV     (-1e9f)
#define EPSV     (1e-16f)

typedef __attribute__((ext_vector_type(16))) __bf16 bf16x16;
typedef __attribute__((ext_vector_type(8)))  float  floatx8;

// ---------------------------------------------------------------- helpers
__device__ __forceinline__ unsigned fkey(float f) {
    unsigned u = __float_as_uint(f);
    return (u & 0x80000000u) ? ~u : (u | 0x80000000u);
}
__device__ __forceinline__ float fdecode(unsigned k) {
    if (k == 0u) return 0.0f;   // untouched segment -> 0 (matches isfinite fallback)
    unsigned u = (k & 0x80000000u) ? (k & 0x7FFFFFFFu) : ~k;
    return __uint_as_float(u);
}
__device__ __forceinline__ float block_reduce_sum(float v, float* sd) {
    int tid = threadIdx.x;
    sd[tid] = v; __syncthreads();
    for (int s = 128; s > 0; s >>= 1) {
        if (tid < s) sd[tid] += sd[tid + s];
        __syncthreads();
    }
    float r = sd[0]; __syncthreads();
    return r;
}

// ---------------------------------------------------------------- pack B into WMMA fragment layout
// In : B f32 row-major [K, N] (ldb = N)
// Out: Bp bf16 [Kpad/32][N/16][32 lanes][16], rows K..Kpad-1 zero-padded.
// One thread per (ktile, ntile, lane) -> writes one 32-byte fragment slice.
__global__ void gnn_packB(const float* __restrict__ B, int N, int K, int Kpad,
                          __bf16* __restrict__ Bp) {
    const int ntiles = N >> 4;
    long i = (long)blockIdx.x * blockDim.x + threadIdx.x;
    long total = (long)(Kpad >> 5) * ntiles * 32;
    if (i >= total) return;
    int lane = (int)(i & 31);
    long t = i >> 5;
    int nt = (int)(t % ntiles);
    int kt = (int)(t / ntiles);
    int col   = nt * 16 + (lane & 15);
    int kbase = kt * 32 + (lane >> 4) * 8;
    bf16x16 v;
    #pragma unroll
    for (int j = 0; j < 8; ++j) {
        int k1 = kbase + j, k2 = kbase + 16 + j;
        v[j]     = (__bf16)((k1 < K) ? B[(long)k1 * N + col] : 0.0f);
        v[j + 8] = (__bf16)((k2 < K) ? B[(long)k2 * N + col] : 0.0f);
    }
    *(bf16x16*)(Bp + i * 16) = v;
}

// ---------------------------------------------------------------- WMMA GEMM (guard-free)
// C[M,N] f32 = A[M,K] f32 * Bp (pre-packed bf16 fragments)
// Requires: M % 16 == 0 per-wave tiles (handled), N % 64 == 0, K % 32 == 0, lda % 4 == 0.
// One wave computes a 16x64 tile; block = 8 waves covering 128 rows.
__global__ __launch_bounds__(256) void gnn_gemm(
    const float* __restrict__ A, int lda,
    const __bf16* __restrict__ Bp,
    float* __restrict__ C, int ldc,
    int M, int N, int K)
{
    const int wave = threadIdx.x >> 5;
    const int lane = threadIdx.x & 31;
    const int row0 = (blockIdx.y * 8 + wave) * 16;
    const int col0 = blockIdx.x * 64;
    if (row0 >= M) return;                 // wave-uniform: EXEC stays all-ones

    floatx8 acc[4];
    #pragma unroll
    for (int t = 0; t < 4; ++t)
        #pragma unroll
        for (int j = 0; j < 8; ++j) acc[t][j] = 0.0f;

    const int lm    = lane & 15;
    const int khalf = (lane >> 4) * 8;     // lanes 16-31 hold K+8 / K+24 halves
    const float* __restrict__ Arow = A + (long)(row0 + lm) * lda + khalf;
    const int ntiles = N >> 4;
    // fragment base for this wave's first column tile at k-tile 0
    const __bf16* __restrict__ bbase = Bp + ((long)(col0 >> 4) * 32 + lane) * 16;
    const long bkstep = (long)ntiles * 512;          // elements per k-tile

    for (int k0 = 0; k0 < K; k0 += 32) {
        const float4 a0 = *(const float4*)(Arow + k0);
        const float4 a1 = *(const float4*)(Arow + k0 + 4);
        const float4 a2 = *(const float4*)(Arow + k0 + 16);
        const float4 a3 = *(const float4*)(Arow + k0 + 20);
        bf16x16 af;
        af[0]  = (__bf16)a0.x; af[1]  = (__bf16)a0.y; af[2]  = (__bf16)a0.z; af[3]  = (__bf16)a0.w;
        af[4]  = (__bf16)a1.x; af[5]  = (__bf16)a1.y; af[6]  = (__bf16)a1.z; af[7]  = (__bf16)a1.w;
        af[8]  = (__bf16)a2.x; af[9]  = (__bf16)a2.y; af[10] = (__bf16)a2.z; af[11] = (__bf16)a2.w;
        af[12] = (__bf16)a3.x; af[13] = (__bf16)a3.y; af[14] = (__bf16)a3.z; af[15] = (__bf16)a3.w;
        const __bf16* bp = bbase + (long)(k0 >> 5) * bkstep;
        #pragma unroll
        for (int nt = 0; nt < 4; ++nt) {
            const bf16x16 bf = *(const bf16x16*)(bp + nt * 512);
            acc[nt] = __builtin_amdgcn_wmma_f32_16x16x32_bf16(
                false, af, false, bf, (short)0, acc[nt], false, false);
        }
    }
    // C/D layout: VGPR r -> row r (lanes 0-15) / row 8+r (lanes 16-31), col = lane&15
    const int rofs = (lane >> 4) * 8;
    #pragma unroll
    for (int nt = 0; nt < 4; ++nt) {
        int cc = col0 + nt * 16 + lm;
        #pragma unroll
        for (int r = 0; r < 8; ++r)
            C[(long)(row0 + rofs + r) * ldc + cc] = acc[nt][r];
    }
}

// ---------------------------------------------------------------- edge-attn table
// attE[t][h] = sum_c (edge_emb[t] @ lin_edge_w[:, h*HID + c]) * att_edge[h][c]
// Collapses the 500k-edge projection to a 3x4 lookup. One block, 256 threads.
__global__ __launch_bounds__(256) void gnn_attE(
    const float* __restrict__ lin_edge_w,  // [HID, HEADS*HID]
    const float* __restrict__ att_edge,    // [HEADS, HID]
    const float* __restrict__ edge_emb,    // [3, HID]
    float* __restrict__ attE)              // [3, HEADS]
{
    __shared__ float wv[HID][HEADS];
    int k = threadIdx.x;
    #pragma unroll
    for (int h = 0; h < HEADS; ++h) {
        float s = 0.0f;
        for (int c = 0; c < HID; ++c)
            s += lin_edge_w[(long)k * (HEADS * HID) + h * HID + c] * att_edge[h * HID + c];
        wv[k][h] = s;
    }
    __syncthreads();
    if (threadIdx.x < 3 * HEADS) {
        int t = threadIdx.x >> 2, h = threadIdx.x & 3;
        float s = 0.0f;
        for (int k2 = 0; k2 < HID; ++k2) s += edge_emb[t * HID + k2] * wv[k2][h];
        attE[t * HEADS + h] = s;
    }
}

// ---------------------------------------------------------------- node combine
__global__ __launch_bounds__(256) void gnn_combine(
    const float* __restrict__ protpre, const float* __restrict__ ligpre,
    const float* __restrict__ prot_b, const float* __restrict__ prot_g, const float* __restrict__ prot_bb,
    const float* __restrict__ lig_b,  const float* __restrict__ lig_g,  const float* __restrict__ lig_bb,
    const float* __restrict__ node_emb, const int* __restrict__ node_type,
    float* __restrict__ h)
{
    __shared__ float sd[256];
    long n = blockIdx.x; int c = threadIdx.x;
    float pv = protpre[n * HID + c] + prot_b[c];
    float lv = ligpre [n * HID + c] + lig_b[c];
    float pm = block_reduce_sum(pv, sd) * (1.0f / HID);
    float pd = pv - pm;
    float pvar = block_reduce_sum(pd * pd, sd) * (1.0f / HID);
    float p = pd * rsqrtf(pvar + 1e-5f) * prot_g[c] + prot_bb[c];
    p = p > 0.0f ? p : 0.0f;
    float lmn = block_reduce_sum(lv, sd) * (1.0f / HID);
    float ld = lv - lmn;
    float lvar = block_reduce_sum(ld * ld, sd) * (1.0f / HID);
    float l = ld * rsqrtf(lvar + 1e-5f) * lig_g[c] + lig_bb[c];
    l = l > 0.0f ? l : 0.0f;
    int t = node_type[n];
    h[n * HID + c] = (t == 0 ? p : l) + node_emb[t * HID + c];
}

// ---------------------------------------------------------------- layernorm
__global__ __launch_bounds__(256) void gnn_ln(
    const float* __restrict__ h, const float* __restrict__ g, const float* __restrict__ b,
    float* __restrict__ z)
{
    __shared__ float sd[256];
    long n = blockIdx.x; int c = threadIdx.x;
    float v = h[n * HID + c];
    float m = block_reduce_sum(v, sd) * (1.0f / HID);
    float d = v - m;
    float var = block_reduce_sum(d * d, sd) * (1.0f / HID);
    z[n * HID + c] = d * rsqrtf(var + 1e-5f) * g[c] + b[c];
}

// ---------------------------------------------------------------- per-node attn dots
// one wave per (node, head): asrc/adst = sum_c xw[n,h,c] * att_{src,dst}[h,c]
__global__ __launch_bounds__(256) void gnn_node_att(
    const float* __restrict__ xw, const float* __restrict__ att_src,
    const float* __restrict__ att_dst, float* __restrict__ asrc, float* __restrict__ adst)
{
    long w = (long)blockIdx.x * 8 + (threadIdx.x >> 5);
    int lane = threadIdx.x & 31;
    if (w >= (long)NNODES * HEADS) return;
    long n = w >> 2; int h = (int)(w & 3);
    float s1 = 0.0f, s2 = 0.0f;
    for (int c = lane; c < HID; c += 32) {
        float v = xw[n * (HEADS * HID) + h * HID + c];
        s1 += v * att_src[h * HID + c];
        s2 += v * att_dst[h * HID + c];
    }
    for (int o = 16; o > 0; o >>= 1) { s1 += __shfl_down(s1, o, 32); s2 += __shfl_down(s2, o, 32); }
    if (lane == 0) { asrc[w] = s1; adst[w] = s2; }
}

// ---------------------------------------------------------------- edge attn + seg-max
__global__ void gnn_edge_attn(
    const float* __restrict__ asrc, const float* __restrict__ adst,
    const float* __restrict__ attE, const int* __restrict__ src, const int* __restrict__ dst,
    const int* __restrict__ etype, float* __restrict__ abuf, unsigned* __restrict__ mkey)
{
    long i = (long)blockIdx.x * blockDim.x + threadIdx.x;
    if (i >= (long)NEDGES * HEADS) return;
    long e = i >> 2; int h = (int)(i & 3);
    int s = src[e], d = dst[e], t = etype[e];
    float a = asrc[(long)s * HEADS + h] + adst[(long)d * HEADS + h] + attE[t * HEADS + h];
    a = a > 0.0f ? a : 0.2f * a;                      // leaky_relu(0.2)
    abuf[i] = a;
    atomicMax(&mkey[(long)d * HEADS + h], fkey(a));
}

// ---------------------------------------------------------------- edge exp + seg-sum
__global__ void gnn_edge_exp(
    const float* __restrict__ abuf, const unsigned* __restrict__ mkey,
    const int* __restrict__ dst, float* __restrict__ eabuf, float* __restrict__ denom)
{
    long i = (long)blockIdx.x * blockDim.x + threadIdx.x;
    if (i >= (long)NEDGES * HEADS) return;
    long e = i >> 2; int h = (int)(i & 3);
    int d = dst[e];
    float ea = expf(abuf[i] - fdecode(mkey[(long)d * HEADS + h]));
    eabuf[i] = ea;
    atomicAdd(&denom[(long)d * HEADS + h], ea);
}

// ---------------------------------------------------------------- aggregation
// one wave per edge; folds mean-over-heads (0.25) into alpha.
__global__ __launch_bounds__(256) void gnn_edge_aggr(
    const float* __restrict__ eabuf, const float* __restrict__ denom,
    const float* __restrict__ xw, const int* __restrict__ src, const int* __restrict__ dst,
    float* __restrict__ out)
{
    long e = (long)blockIdx.x * 8 + (threadIdx.x >> 5);
    int lane = threadIdx.x & 31;
    if (e >= NEDGES) return;
    int s = src[e], d = dst[e];
    float al[HEADS];
    #pragma unroll
    for (int h = 0; h < HEADS; ++h)
        al[h] = 0.25f * eabuf[e * HEADS + h] / (denom[(long)d * HEADS + h] + EPSV);
    for (int c = lane; c < HID; c += 32) {
        float v = 0.0f;
        #pragma unroll
        for (int h = 0; h < HEADS; ++h) v += al[h] * xw[(long)s * (HEADS * HID) + h * HID + c];
        atomicAdd(&out[(long)d * HID + c], v);
    }
}

// ---------------------------------------------------------------- residual
__global__ void gnn_residual(const float* __restrict__ out, const float* __restrict__ bias,
                             float* __restrict__ h)
{
    long i = (long)blockIdx.x * blockDim.x + threadIdx.x;
    if (i >= (long)NNODES * HID) return;
    float v = out[i] + bias[i & (HID - 1)];
    h[i] += v > 0.0f ? v : 0.0f;
}

// ---------------------------------------------------------------- pooling logits
// one wave per node: logits = tanh(t + b1) @ w2 + b2, mask, seg-max per graph
__global__ __launch_bounds__(256) void gnn_logits(
    const float* __restrict__ tbuf, const float* __restrict__ b1,
    const float* __restrict__ w2, const float* __restrict__ b2,
    const int* __restrict__ node_type, const int* __restrict__ batch,
    float* __restrict__ logits, unsigned* __restrict__ gmkey)
{
    long n = (long)blockIdx.x * 8 + (threadIdx.x >> 5);
    int lane = threadIdx.x & 31;
    if (n >= NNODES) return;
    float acc[ATTH] = {0.f, 0.f, 0.f, 0.f};
    for (int j = lane; j < ATTD; j += 32) {
        float tv = tanhf(tbuf[n * ATTD + j] + b1[j]);
        #pragma unroll
        for (int h = 0; h < ATTH; ++h) acc[h] += tv * w2[j * ATTH + h];
    }
    for (int o = 16; o > 0; o >>= 1)
        #pragma unroll
        for (int h = 0; h < ATTH; ++h) acc[h] += __shfl_down(acc[h], o, 32);
    if (lane == 0) {
        bool prot = (node_type[n] == 0);
        int g = batch[n];
        #pragma unroll
        for (int h = 0; h < ATTH; ++h) {
            float L = prot ? (acc[h] + b2[h]) : NEGV;
            logits[n * ATTH + h] = L;
            atomicMax(&gmkey[g * ATTH + h], fkey(L));
        }
    }
}

// ---------------------------------------------------------------- graph softmax sum
__global__ void gnn_pool_sum(const float* __restrict__ logits, const unsigned* __restrict__ gmkey,
                             const int* __restrict__ batch, float* __restrict__ elbuf,
                             float* __restrict__ sbuf)
{
    long i = (long)blockIdx.x * blockDim.x + threadIdx.x;
    if (i >= (long)NNODES * ATTH) return;
    long n = i >> 2; int h = (int)(i & 3);
    int g = batch[n];
    float el = expf(logits[i] - fdecode(gmkey[g * ATTH + h]));
    elbuf[i] = el;
    atomicAdd(&sbuf[g * ATTH + h], el);
}

// ---------------------------------------------------------------- pooled output
__global__ __launch_bounds__(256) void gnn_pool_agg(
    const float* __restrict__ elbuf, const float* __restrict__ sbuf,
    const float* __restrict__ h, const int* __restrict__ node_type,
    const int* __restrict__ batch, float* __restrict__ dout)
{
    long n = (long)blockIdx.x * 8 + (threadIdx.x >> 5);
    int lane = threadIdx.x & 31;
    if (n >= NNODES) return;
    if (node_type[n] != 0) return;          // px == 0 for ligand nodes
    int g = batch[n];
    float sc[ATTH];
    #pragma unroll
    for (int hh = 0; hh < ATTH; ++hh)
        sc[hh] = elbuf[n * ATTH + hh] / (sbuf[g * ATTH + hh] + EPSV);
    for (int c = lane; c < HID; c += 32) {
        float hv = h[n * HID + c];
        #pragma unroll
        for (int hh = 0; hh < ATTH; ++hh)
            atomicAdd(&dout[(long)g * (ATTH * HID) + hh * HID + c], sc[hh] * hv);
    }
}

// ================================================================ host
extern "C" void kernel_launch(void* const* d_in, const int* in_sizes, int n_in,
                              void* d_out, int out_size, void* d_ws, size_t ws_size,
                              hipStream_t stream) {
    (void)in_sizes; (void)n_in; (void)out_size; (void)ws_size;
    // setup_inputs() dict insertion order, recursively flattened:
    const float* x        = (const float*)d_in[0];
    const float* prot_w   = (const float*)d_in[1];
    const float* prot_b   = (const float*)d_in[2];
    const float* prot_lng = (const float*)d_in[3];
    const float* prot_lnb = (const float*)d_in[4];
    const float* lig_w    = (const float*)d_in[5];
    const float* lig_b    = (const float*)d_in[6];
    const float* lig_lng  = (const float*)d_in[7];
    const float* lig_lnb  = (const float*)d_in[8];
    const float* node_emb = (const float*)d_in[9];
    const float* edge_emb = (const float*)d_in[10];
    // layers: 11 + 8*L : lin_w, lin_edge_w, att_src, att_dst, att_edge, bias, ln_g, ln_b
    const float* w1 = (const float*)d_in[35];
    const float* b1 = (const float*)d_in[36];
    const float* w2 = (const float*)d_in[37];
    const float* b2 = (const float*)d_in[38];
    const int* edge_index = (const int*)d_in[39];
    const int* etype      = (const int*)d_in[40];
    const int* node_type  = (const int*)d_in[41];
    const int* batch      = (const int*)d_in[42];
    const int* esrc = edge_index;
    const int* edst = edge_index + NEDGES;
    float* dout = (float*)d_out;

    // workspace layout
    char* ws = (char*)d_ws;
    size_t off = 0;
    auto take = [&](size_t bytes) { size_t r = off; off += (bytes + 255) & ~(size_t)255; return r; };
    float*    hbuf  = (float*)(ws + take((size_t)NNODES * HID * 4));
    float*    zbuf  = (float*)(ws + take((size_t)NNODES * HID * 4));   // z, then reused as aggregation out
    float*    xwbuf = (float*)(ws + take((size_t)NNODES * HEADS * HID * 4)); // xw / protpre+ligpre / tbuf
    float*    abuf  = (float*)(ws + take((size_t)NEDGES * HEADS * 4)); // edge a, reused as logits
    float*    eabuf = (float*)(ws + take((size_t)NEDGES * HEADS * 4)); // edge exp, reused as el
    float*    asrcb = (float*)(ws + take((size_t)NNODES * HEADS * 4));
    float*    adstb = (float*)(ws + take((size_t)NNODES * HEADS * 4));
    unsigned* mkey  = (unsigned*)(ws + take((size_t)NNODES * HEADS * 4)); // reused as gmkey
    float*    denom = (float*)(ws + take((size_t)NNODES * HEADS * 4));    // reused as sbuf
    float*    attE  = (float*)(ws + take(3 * 16 * 4));                    // 3 layers x [3x4]
    __bf16* wprot = (__bf16*)(ws + take((size_t)ESM * HID * 2));          // packed fragments
    __bf16* wlig  = (__bf16*)(ws + take((size_t)LIGK * HID * 2));
    __bf16* wlin[3];
    for (int L = 0; L < 3; ++L) wlin[L] = (__bf16*)(ws + take((size_t)HID * HEADS * HID * 2));
    __bf16* w1b = (__bf16*)(ws + take((size_t)HID * ATTD * 2));

    // 1) pack weights into WMMA fragment layout (bf16), amortized over ~391 row-blocks
    {
        long tp = (long)(ESM / 32) * (HID / 16) * 32;
        gnn_packB<<<(unsigned)((tp + 255) / 256), 256, 0, stream>>>(prot_w, HID, ESM, ESM, wprot);
        long tl = (long)(LIGK / 32) * (HID / 16) * 32;
        gnn_packB<<<(unsigned)((tl + 255) / 256), 256, 0, stream>>>(lig_w, HID, LIGD, LIGK, wlig);
        long tw = (long)(HID / 32) * ((HEADS * HID) / 16) * 32;
        for (int L = 0; L < 3; ++L)
            gnn_packB<<<(unsigned)((tw + 255) / 256), 256, 0, stream>>>(
                (const float*)d_in[11 + 8 * L], HEADS * HID, HID, HID, wlin[L]);
        long t1 = (long)(HID / 32) * (ATTD / 16) * 32;
        gnn_packB<<<(unsigned)((t1 + 255) / 256), 256, 0, stream>>>(w1, ATTD, HID, HID, w1b);
    }

    // 2) collapse edge projection to 3x4 tables
    for (int L = 0; L < 3; ++L)
        gnn_attE<<<1, 256, 0, stream>>>((const float*)d_in[11 + 8 * L + 1],
                                        (const float*)d_in[11 + 8 * L + 4],
                                        edge_emb, attE + L * 16);

    // 3) input projections (WMMA GEMMs) -> protpre/ligpre in xwbuf
    float* protpre = xwbuf;
    float* ligpre  = xwbuf + (size_t)NNODES * HID;
    {
        dim3 g1(HID / 64, (NNODES + 127) / 128);
        gnn_gemm<<<g1, 256, 0, stream>>>(x, ESM, wprot, protpre, HID, NNODES, HID, ESM);
        // ligand: K padded to 64; x cols 36..63 are multiplied by zero B rows
        gnn_gemm<<<g1, 256, 0, stream>>>(x, ESM, wlig, ligpre, HID, NNODES, HID, LIGK);
    }
    gnn_combine<<<NNODES, 256, 0, stream>>>(protpre, ligpre, prot_b, prot_lng, prot_lnb,
                                            lig_b, lig_lng, lig_lnb, node_emb, node_type, hbuf);

    // 4) GAT layers
    for (int L = 0; L < 3; ++L) {
        const float* att_src = (const float*)d_in[11 + 8 * L + 2];
        const float* att_dst = (const float*)d_in[11 + 8 * L + 3];
        const float* bias    = (const float*)d_in[11 + 8 * L + 5];
        const float* ln_g    = (const float*)d_in[11 + 8 * L + 6];
        const float* ln_b    = (const float*)d_in[11 + 8 * L + 7];

        gnn_ln<<<NNODES, 256, 0, stream>>>(hbuf, ln_g, ln_b, zbuf);
        dim3 g2((HEADS * HID) / 64, (NNODES + 127) / 128);
        gnn_gemm<<<g2, 256, 0, stream>>>(zbuf, HID, wlin[L], xwbuf, HEADS * HID,
                                         NNODES, HEADS * HID, HID);
        gnn_node_att<<<(NNODES * HEADS + 7) / 8, 256, 0, stream>>>(xwbuf, att_src, att_dst,
                                                                   asrcb, adstb);
        hipMemsetAsync(mkey, 0, (size_t)NNODES * HEADS * 4, stream);   // key 0 == -inf sentinel
        hipMemsetAsync(denom, 0, (size_t)NNODES * HEADS * 4, stream);
        hipMemsetAsync(zbuf, 0, (size_t)NNODES * HID * 4, stream);     // reuse z as agg out
        long ne4 = (long)NEDGES * HEADS;
        gnn_edge_attn<<<(unsigned)((ne4 + 255) / 256), 256, 0, stream>>>(
            asrcb, adstb, attE + L * 16, esrc, edst, etype, abuf, mkey);
        gnn_edge_exp<<<(unsigned)((ne4 + 255) / 256), 256, 0, stream>>>(
            abuf, mkey, edst, eabuf, denom);
        gnn_edge_aggr<<<(NEDGES + 7) / 8, 256, 0, stream>>>(eabuf, denom, xwbuf, esrc, edst, zbuf);
        gnn_residual<<<(NNODES * HID + 255) / 256, 256, 0, stream>>>(zbuf, bias, hbuf);
    }

    // 5) attention pooling
    float* tbuf = xwbuf;  // [N, ATTD]
    {
        dim3 g3(ATTD / 64, (NNODES + 127) / 128);
        gnn_gemm<<<g3, 256, 0, stream>>>(hbuf, HID, w1b, tbuf, ATTD, NNODES, ATTD, HID);
    }
    hipMemsetAsync(mkey, 0, (size_t)NGRAPHS * ATTH * 4, stream);   // gmkey
    hipMemsetAsync(denom, 0, (size_t)NGRAPHS * ATTH * 4, stream);  // sbuf
    hipMemsetAsync(dout, 0, (size_t)NGRAPHS * ATTH * HID * 4, stream);
    float* logits = abuf;
    float* elbuf  = eabuf;
    gnn_logits<<<(NNODES + 7) / 8, 256, 0, stream>>>(tbuf, b1, w2, b2, node_type, batch,
                                                     logits, mkey);
    gnn_pool_sum<<<(NNODES * ATTH + 255) / 256, 256, 0, stream>>>(logits, mkey, batch,
                                                                  elbuf, denom);
    gnn_pool_agg<<<(NNODES + 7) / 8, 256, 0, stream>>>(elbuf, denom, hbuf, node_type,
                                                       batch, dout);
}
